// DCP_84026740179147
// MI455X (gfx1250) — compile-verified, matrix operand order checked by reading, not run on hardware
//
#include <hip/hip_runtime.h>

// ---------------- problem constants ----------------
#define BATCH   32
#define IMSZ    (512 * 512)          // 262144 pixels per channel plane
#define NUMPX   262                  // IMSZ / 1000
#define NBINS   1024                 // dark-channel histogram bins
#define CHUNKS  32                   // blocks per batch for streaming passes
#define PIXPB   (IMSZ / CHUNKS)      // 8192 pixels per block
#define THREADS 256                  // 8 waves of 32
#define NWAVES  (THREADS / 32)

// ---------------- workspace layout (32-bit words) ----------------
#define W_HIST   0                                   // BATCH*NBINS u32
#define W_THRESH (BATCH * NBINS)                     // BATCH int
#define W_REM    (W_THRESH + BATCH)                  // BATCH int
#define W_TIE    (W_REM + BATCH)                     // BATCH u32 (atomic tie slots)
#define W_PART   (W_TIE + BATCH)                     // BATCH*CHUNKS*NWAVES*3 float
#define W_AVAL   (W_PART + BATCH * CHUNKS * NWAVES * 3)  // BATCH*3 float
#define W_AINV   (W_AVAL + BATCH * 3)                // BATCH*3 float
#define W_ZERO_END W_PART                            // words that must be zeroed per launch

typedef __attribute__((ext_vector_type(2)))  float    v2f;
typedef __attribute__((ext_vector_type(4)))  float    v4f;   // native vector: OK for NT store
typedef __attribute__((ext_vector_type(8)))  float    v8f;
typedef __attribute__((ext_vector_type(16))) _Float16 v16h;

// ---------------- wave32 sum via WMMA (A = ones -> column sums) ----------------
__device__ __forceinline__ float wave_sum32(float x) {
#if __has_builtin(__builtin_amdgcn_wmma_f32_16x16x4_f32)
  // f32 WMMA: A(16x4)=ones, B(4x16): VGPR0 carries rows K=0 (lanes 0-15) and
  // K=2 (lanes 16-31), VGPR1 zeroed => D[i][j] = p(j) + p(j+16) for all i.
  v2f a; a[0] = 1.0f; a[1] = 1.0f;
  v2f b; b[0] = x;    b[1] = 0.0f;
  v8f c = {};
  v8f d = __builtin_amdgcn_wmma_f32_16x16x4_f32(false, a, false, b, (short)0, c,
                                                false, false);
  float s = d[0];
#elif __has_builtin(__builtin_amdgcn_wmma_f32_16x16x32_f16)
  // fallback: f16 WMMA, same column-sum structure (B element 0 = K=0/K=8 rows)
  v16h a, b;
#pragma unroll
  for (int i = 0; i < 16; ++i) { a[i] = (_Float16)1.0f; b[i] = (_Float16)0.0f; }
  b[0] = (_Float16)x;
  v8f c = {};
  v8f d = __builtin_amdgcn_wmma_f32_16x16x32_f16(false, a, false, b, (short)0, c,
                                                 false, false);
  float s = d[0];
#else
  float s = x;
  s += __shfl_xor(s, 16, 32);
#endif
  // lanes now hold pairwise column sums; butterfly the remaining 16 columns
  s += __shfl_xor(s, 1, 32);
  s += __shfl_xor(s, 2, 32);
  s += __shfl_xor(s, 4, 32);
  s += __shfl_xor(s, 8, 32);
  return s;  // every lane holds the full 32-lane sum
}

__device__ __forceinline__ int dark_bin(float r, float g, float b) {
  float d = fminf(fminf(r, g), b);
  int bin = (int)(d * (float)NBINS);
  bin = bin < 0 ? 0 : (bin > NBINS - 1 ? NBINS - 1 : bin);
  return bin;
}

// ---------------- pass 0: zero histogram / counters ----------------
__global__ void k_init(unsigned* __restrict__ ws) {
  int i = blockIdx.x * blockDim.x + threadIdx.x;
  if (i < W_ZERO_END) ws[i] = 0u;
}

// ---------------- pass 1: per-batch dark-channel histogram ----------------
__global__ void k_hist(const float* __restrict__ img, unsigned* __restrict__ ws) {
  __shared__ unsigned lhist[NBINS];
  const int b = blockIdx.y, chunk = blockIdx.x;
  for (int i = threadIdx.x; i < NBINS; i += THREADS) lhist[i] = 0u;
  __syncthreads();

  const float* pr = img + (size_t)b * 3 * IMSZ;
  const float* pg = pr + IMSZ;
  const float* pb = pg + IMSZ;
  const int base = chunk * PIXPB;

  for (int i = threadIdx.x * 4; i < PIXPB; i += THREADS * 4) {
    const int p = base + i;
    __builtin_prefetch(pr + p + THREADS * 4, 0, 0);
    v4f r = *(const v4f*)(pr + p);
    v4f g = *(const v4f*)(pg + p);
    v4f u = *(const v4f*)(pb + p);
#pragma unroll
    for (int k = 0; k < 4; ++k)
      atomicAdd(&lhist[dark_bin(r[k], g[k], u[k])], 1u);
  }
  __syncthreads();

  unsigned* gh = ws + W_HIST + b * NBINS;
  for (int i = threadIdx.x; i < NBINS; i += THREADS) {
    unsigned v = lhist[i];
    if (v) atomicAdd(&gh[i], v);
  }
}

// ---------------- pass 2: threshold bin via suffix scan ----------------
__global__ void k_thresh(unsigned* __restrict__ ws) {
  int b = threadIdx.x;
  if (b >= BATCH) return;
  const unsigned* h = ws + W_HIST + b * NBINS;
  unsigned acc = 0;
  int T = 0;
  for (int bin = NBINS - 1; bin >= 0; --bin) {
    acc += h[bin];
    if (acc >= NUMPX) { T = bin; break; }
  }
  ((int*)ws)[W_THRESH + b] = T;
  ((int*)ws)[W_REM + b] = NUMPX - (int)(acc - h[T]);  // how many to take from bin T
}

// ---------------- pass 3: gather top pixels, WMMA wave reduction ----------------
__device__ __forceinline__ void select_px(float r, float g, float u, int T, int rem,
                                          unsigned* tieCnt, float& sr, float& sg,
                                          float& sb) {
  int bin = dark_bin(r, g, u);
  bool sel = bin > T;
  if (bin == T) {
    unsigned slot = atomicAdd(tieCnt, 1u);
    sel = slot < (unsigned)rem;
  }
  if (sel) { sr += r; sg += g; sb += u; }
}

__global__ void k_gather(const float* __restrict__ img, unsigned* __restrict__ ws) {
  const int b = blockIdx.y, chunk = blockIdx.x;
  const int T = ((const int*)ws)[W_THRESH + b];
  const int rem = ((const int*)ws)[W_REM + b];
  unsigned* tieCnt = ws + W_TIE + b;

  const float* pr = img + (size_t)b * 3 * IMSZ;
  const float* pg = pr + IMSZ;
  const float* pb = pg + IMSZ;
  const int base = chunk * PIXPB;

  float sr = 0.f, sg = 0.f, sb = 0.f;
  for (int i = threadIdx.x * 4; i < PIXPB; i += THREADS * 4) {
    const int p = base + i;
    v4f r = *(const v4f*)(pr + p);
    v4f g = *(const v4f*)(pg + p);
    v4f u = *(const v4f*)(pb + p);
#pragma unroll
    for (int k = 0; k < 4; ++k)
      select_px(r[k], g[k], u[k], T, rem, tieCnt, sr, sg, sb);
  }

  // EXEC is all-ones here (no divergence): WMMA wave reductions, one per channel
  float tr = wave_sum32(sr);
  float tg = wave_sum32(sg);
  float tb = wave_sum32(sb);

  const int wave = threadIdx.x >> 5;
  if ((threadIdx.x & 31) == 0) {
    float* part = (float*)ws + W_PART + (((b * CHUNKS) + chunk) * NWAVES + wave) * 3;
    part[0] = tr;
    part[1] = tg;
    part[2] = tb;
  }
}

// ---------------- pass 4: fixed-order reduce -> a, 1/a ----------------
__global__ void k_final_a(unsigned* __restrict__ ws) {
  int i = threadIdx.x;
  if (i >= BATCH * 3) return;
  const int b = i / 3, c = i % 3;
  const float* part = (const float*)ws + W_PART;
  float s = 0.f;
  for (int j = 0; j < CHUNKS * NWAVES; ++j)
    s += part[((b * CHUNKS * NWAVES) + j) * 3 + c];
  float a = s / (float)NUMPX + 1e-6f;
  ((float*)ws)[W_AVAL + i] = a;
  ((float*)ws)[W_AINV + i] = 1.0f / a;
}

// ---------------- pass 5: fused transmission + dehaze ----------------
__global__ void k_dehaze(const float* __restrict__ img,
                         const unsigned* __restrict__ ws,
                         float* __restrict__ out) {
  const int b = blockIdx.y, chunk = blockIdx.x;
  const float* aval = (const float*)ws + W_AVAL + b * 3;
  const float* ainv = (const float*)ws + W_AINV + b * 3;
  const float ar = aval[0], ag = aval[1], ab = aval[2];
  const float ir = ainv[0], ig = ainv[1], ib = ainv[2];

  const float* pr = img + (size_t)b * 3 * IMSZ;
  const float* pg = pr + IMSZ;
  const float* pb = pg + IMSZ;
  float* qr = out + (size_t)b * 3 * IMSZ;
  float* qg = qr + IMSZ;
  float* qb = qg + IMSZ;
  const int base = chunk * PIXPB;

  for (int i = threadIdx.x * 4; i < PIXPB; i += THREADS * 4) {
    const int p = base + i;
    __builtin_prefetch(pr + p + THREADS * 4, 0, 0);
    __builtin_prefetch(pg + p + THREADS * 4, 0, 0);
    __builtin_prefetch(pb + p + THREADS * 4, 0, 0);
    v4f r = *(const v4f*)(pr + p);
    v4f g = *(const v4f*)(pg + p);
    v4f u = *(const v4f*)(pb + p);
    v4f orr, og, ob;
#pragma unroll
    for (int k = 0; k < 4; ++k) {
      float rv = r[k], gv = g[k], uv = u[k];
      float m = fminf(fminf(rv * ir, gv * ig), uv * ib);
      float t = fmaxf(1.0f - 0.75f * m, 0.1f);
      float q = 1.0f / t;
      orr[k] = (rv - ar) * q + ar;
      og[k]  = (gv - ag) * q + ag;
      ob[k]  = (uv - ab) * q + ab;
    }
    // non-temporal: keep img resident in L2 instead of the output
    __builtin_nontemporal_store(orr, (v4f*)(qr + p));
    __builtin_nontemporal_store(og,  (v4f*)(qg + p));
    __builtin_nontemporal_store(ob,  (v4f*)(qb + p));
  }
}

// ---------------- launcher ----------------
extern "C" void kernel_launch(void* const* d_in, const int* in_sizes, int n_in,
                              void* d_out, int out_size, void* d_ws, size_t ws_size,
                              hipStream_t stream) {
  const float* img = (const float*)d_in[0];
  float* out = (float*)d_out;
  unsigned* ws = (unsigned*)d_ws;

  k_init<<<(W_ZERO_END + 255) / 256, 256, 0, stream>>>(ws);

  dim3 grid(CHUNKS, BATCH);
  k_hist<<<grid, THREADS, 0, stream>>>(img, ws);
  k_thresh<<<1, 32, 0, stream>>>(ws);
  k_gather<<<grid, THREADS, 0, stream>>>(img, ws);
  k_final_a<<<1, 128, 0, stream>>>(ws);
  k_dehaze<<<grid, THREADS, 0, stream>>>(img, ws, out);
}
